// FFSpikingLayer_62723702391158
// MI455X (gfx1250) — matrix operand order repaired and checked by reading, not compiled, not run on hardware
//
#include <hip/hip_runtime.h>

typedef __attribute__((ext_vector_type(16))) _Float16 v16h;
typedef __attribute__((ext_vector_type(8)))  _Float16 v8h;
typedef __attribute__((ext_vector_type(4)))  _Float16 h4;
typedef __attribute__((ext_vector_type(8)))  float    v8f;
typedef __attribute__((ext_vector_type(4)))  float    f4;

#define T_STEPS 64
#define BATCH   256
#define K_IN    784
#define KP      800          // K padded to 25 chunks of 32 (zero-filled tail)
#define N_OUT   2048
#define TILE_M  32           // batch rows per workgroup
#define TILE_N  64           // out cols per workgroup
#define KCHUNKS (KP / 32)
#define NTHREADS 128         // 4 waves: 2(M) x 2(N), each wave -> 16x32 output

// Fused GEMM + LIF scan.
// Each wave computes a 16x32 C-tile as two v_wmma_f32_16x16x32_f16 fragments
// sharing one A-fragment (halves A LDS traffic per FLOP). Membrane potential v
// and spike counts live in registers across all 64 timesteps; the W tile lives
// in LDS for the whole kernel. Spike/count stores are non-temporal so the
// 136 MB output stream does not evict the L2-resident x/W working set.
__global__ __launch_bounds__(NTHREADS) void
lif_wmma_kernel(const float* __restrict__ x,     // [T,B,K_IN]
                const float* __restrict__ W,     // [N_OUT,K_IN]
                const float* __restrict__ bias,  // [N_OUT]
                float* __restrict__ out)         // spk [T,B,N_OUT] ++ count [B,N_OUT]
{
    extern __shared__ _Float16 smem[];
    _Float16* Wl = smem;                       // [TILE_N][KP] f16
    _Float16* Al = smem + TILE_N * KP;         // [TILE_M][KP] f16

    const int tid    = threadIdx.x;
    const int lane   = tid & 31;
    const int wave   = tid >> 5;
    const int waveM  = wave & 1;               // 0..1
    const int waveN  = wave >> 1;              // 0..1 (each covers 32 cols)
    const int laneN  = lane & 15;
    const int laneHi = lane >> 4;              // K-half select per ISA layout
    const int o0     = blockIdx.x * TILE_N;
    const int b0     = blockIdx.y * TILE_M;

    // ---- Stage W tile into LDS once (float4 -> half4, zero-pad K tail) ----
    // KP/4 = 200 vec4 per row; k4 >= 196 is the zero-filled tail.
    for (int idx = tid; idx < TILE_N * (KP / 4); idx += NTHREADS) {
        int row = idx / (KP / 4), k4 = idx - row * (KP / 4);
        f4 w = {0.f, 0.f, 0.f, 0.f};
        if (k4 < K_IN / 4) w = *(const f4*)(W + (size_t)(o0 + row) * K_IN + k4 * 4);
        h4 h; h.x = (_Float16)w.x; h.y = (_Float16)w.y; h.z = (_Float16)w.z; h.w = (_Float16)w.w;
        *(h4*)(Wl + row * KP + k4 * 4) = h;
    }

    const float bval0 = bias[o0 + waveN * 32 + laneN];
    const float bval1 = bias[o0 + waveN * 32 + 16 + laneN];

    // Per-lane fragment base addresses in LDS (contiguous k = 16*laneHi + s order for A and B):
    // A (16x32): lanes 0-15 hold M=lane, K 0..15; lanes 16-31 K 16..31
    const _Float16* aBase  = Al + (waveM * 16 + laneN) * KP + laneHi * 16;
    // B (32x16): lane n<16 holds col n, K 0..15; lane n+16 holds K 16..31 (32B contig)
    const _Float16* bBase0 = Wl + (waveN * 32 + laneN) * KP + laneHi * 16;
    const _Float16* bBase1 = Wl + (waveN * 32 + 16 + laneN) * KP + laneHi * 16;

    // C/D layout: VGPR r -> row M = r (lanes 0-15) or r+8 (lanes 16-31), col = laneN
    const int gn0 = o0 + waveN * 32 + laneN;
    const int gm0 = b0 + waveM * 16 + laneHi * 8;

    float v0[8], v1[8], cnt0[8], cnt1[8];
#pragma unroll
    for (int r = 0; r < 8; ++r) { v0[r] = v1[r] = 0.0f; cnt0[r] = cnt1[r] = 0.0f; }

    float* spk   = out;
    float* count = out + (size_t)T_STEPS * BATCH * N_OUT;

    for (int t = 0; t < T_STEPS; ++t) {
        __syncthreads();   // previous iteration's A reads complete before refill
        // ---- Stage x[t] tile into LDS (float4 -> half4, zero-pad K tail) ----
        const float* xt = x + ((size_t)t * BATCH + b0) * K_IN;
        for (int idx = tid; idx < TILE_M * (KP / 4); idx += NTHREADS) {
            int row = idx / (KP / 4), k4 = idx - row * (KP / 4);
            f4 a = {0.f, 0.f, 0.f, 0.f};
            if (k4 < K_IN / 4) a = *(const f4*)(xt + (size_t)row * K_IN + k4 * 4);
            h4 h; h.x = (_Float16)a.x; h.y = (_Float16)a.y; h.z = (_Float16)a.z; h.w = (_Float16)a.w;
            *(h4*)(Al + row * KP + k4 * 4) = h;
        }
        // Prefetch next timestep's x tile toward L2 (global_prefetch_b8)
        if (t + 1 < T_STEPS) {
            const char* pf = (const char*)(x + ((size_t)(t + 1) * BATCH + b0) * K_IN);
            __builtin_prefetch(pf + (size_t)tid * 256, 0, 1);
        }
        __syncthreads();

        // ---- 16x32 GEMM tile: 25 K-chunks, 2 WMMAs per chunk sharing A ----
        v8f acc0 = {}, acc1 = {};
#pragma unroll 5
        for (int c = 0; c < KCHUNKS; ++c) {
            const _Float16* ap = aBase + c * 32;
            v8h alo = *(const v8h*)(ap);          // ds_load_b128: K lo-quad
            v8h ahi = *(const v8h*)(ap + 8);      // ds_load_b128: K hi-quad
            v16h afrag;
#pragma unroll
            for (int i = 0; i < 8; ++i) { afrag[i] = alo[i]; afrag[i + 8] = ahi[i]; }
            v16h bfrag0 = *(const v16h*)(bBase0 + c * 32);  // 32B contiguous
            v16h bfrag1 = *(const v16h*)(bBase1 + c * 32);
            acc0 = __builtin_amdgcn_wmma_f32_16x16x32_f16(
                false, afrag, false, bfrag0, (short)0, acc0, false, false);
            acc1 = __builtin_amdgcn_wmma_f32_16x16x32_f16(
                false, afrag, false, bfrag1, (short)0, acc1, false, false);
        }

        // ---- LIF update: v=(v+cur)/2; spike -> hard reset; NT spike store ----
        size_t rowBase = ((size_t)t * BATCH + gm0) * N_OUT + gn0;
#pragma unroll
        for (int r = 0; r < 8; ++r) {
            float cur  = acc0[r] + bval0;
            float vn   = 0.5f * (v0[r] + cur);
            bool  fire = (vn >= 1.0f);
            float s    = fire ? 1.0f : 0.0f;
            cnt0[r] += s;
            v0[r]    = fire ? 0.0f : vn;
            __builtin_nontemporal_store(s, &spk[rowBase + (size_t)r * N_OUT]);

            float cur1  = acc1[r] + bval1;
            float vn1   = 0.5f * (v1[r] + cur1);
            bool  fire1 = (vn1 >= 1.0f);
            float s1    = fire1 ? 1.0f : 0.0f;
            cnt1[r] += s1;
            v1[r]    = fire1 ? 0.0f : vn1;
            __builtin_nontemporal_store(s1, &spk[rowBase + (size_t)r * N_OUT + 16]);
        }
    }

    // ---- Spike counts (non-temporal: pure streaming output) ----
    size_t cBase = (size_t)gm0 * N_OUT + gn0;
#pragma unroll
    for (int r = 0; r < 8; ++r) {
        __builtin_nontemporal_store(cnt0[r], &count[cBase + (size_t)r * N_OUT]);
        __builtin_nontemporal_store(cnt1[r], &count[cBase + (size_t)r * N_OUT + 16]);
    }
}

extern "C" void kernel_launch(void* const* d_in, const int* in_sizes, int n_in,
                              void* d_out, int out_size, void* d_ws, size_t ws_size,
                              hipStream_t stream) {
    (void)in_sizes; (void)n_in; (void)out_size; (void)d_ws; (void)ws_size;
    const float* x = (const float*)d_in[0];
    const float* W = (const float*)d_in[1];
    const float* b = (const float*)d_in[2];
    float* out = (float*)d_out;

    const size_t lds_bytes = (size_t)(TILE_M + TILE_N) * KP * sizeof(_Float16); // 153.6 KB
    // Opt in to >64KB dynamic LDS (WGP supports 320KB; 2 WGs co-resident).
    (void)hipFuncSetAttribute((const void*)lif_wmma_kernel,
                              hipFuncAttributeMaxDynamicSharedMemorySize,
                              (int)lds_bytes);

    dim3 grid(N_OUT / TILE_N, BATCH / TILE_M);  // (32, 8) = 256 workgroups
    hipLaunchKernelGGL(lif_wmma_kernel, grid, dim3(NTHREADS), lds_bytes, stream,
                       x, W, b, out);
}
